// FastQAAnswerModule_29798483100476
// MI455X (gfx1250) — matrix operand order repaired and checked by reading, not compiled, not run on hardware
//
#include <hip/hip_runtime.h>
#include <hip/hip_bf16.h>

#define BB 64
#define LQN 64
#define LSN 512
#define SN 512
#define NEGV (-1e6f)

typedef __attribute__((ext_vector_type(16))) __bf16 v16bf;
typedef __attribute__((ext_vector_type(8)))  __bf16 v8bf;
typedef __attribute__((ext_vector_type(8)))  float  v8f;

// ---------------------------------------------------------------------------
// Kernel 1: question attention pooling -> q_state [B,S]
// ---------------------------------------------------------------------------
__global__ __launch_bounds__(256) void qa_qpool(
    const float* __restrict__ eq, const int* __restrict__ qlen,
    const float* __restrict__ W_qa, float* __restrict__ q_state)
{
    __shared__ float att[LQN];
    __shared__ float wts[LQN];
    const int b = blockIdx.x;
    const int t = threadIdx.x;

    // att[q] = eq[b,q,:] . W_qa   (4 threads per q)
    {
        const int q = t >> 2, l4 = t & 3;
        const float* row = eq + ((size_t)b * LQN + q) * SN;
        float p = 0.f;
        for (int j = l4; j < SN; j += 4) p = fmaf(row[j], W_qa[j], p);
        p += __shfl_xor(p, 1);
        p += __shfl_xor(p, 2);
        if (l4 == 0) att[q] = p;
    }
    __syncthreads();

    // masked softmax over 64 positions (wave 0)
    if (t < 32) {
        const int ql = qlen[b];
        float v0 = att[t]      + ((t      >= ql) ? NEGV : 0.f);
        float v1 = att[t + 32] + ((t + 32 >= ql) ? NEGV : 0.f);
        float m = fmaxf(v0, v1);
        for (int off = 16; off > 0; off >>= 1) m = fmaxf(m, __shfl_xor(m, off));
        float e0 = __expf(v0 - m), e1 = __expf(v1 - m);
        float s = e0 + e1;
        for (int off = 16; off > 0; off >>= 1) s += __shfl_xor(s, off);
        const float inv = 1.f / s;
        wts[t]      = e0 * inv;
        wts[t + 32] = e1 * inv;
    }
    __syncthreads();

    // q_state[d] = sum_q eq[b,q,d] * w[q]
    for (int d = t; d < SN; d += 256) {
        float acc = 0.f;
        for (int q = 0; q < LQN; ++q)
            acc = fmaf(eq[((size_t)b * LQN + q) * SN + d], wts[q], acc);
        q_state[(size_t)b * SN + d] = acc;
    }
}

// ---------------------------------------------------------------------------
// Kernel 2: q_start_inter / q_end_inter = q_state @ W + bias
// ---------------------------------------------------------------------------
__global__ __launch_bounds__(256) void qa_inter(
    const float* __restrict__ q_state,
    const float* __restrict__ W_qsi, const float* __restrict__ b_qsi,
    const float* __restrict__ W_qei, const float* __restrict__ b_qei,
    float* __restrict__ inter_s, float* __restrict__ inter_e)
{
    const int b = blockIdx.x;
    const int which = blockIdx.y;
    const int t = threadIdx.x;
    const float* W    = which ? W_qei : W_qsi;
    const float* bias = which ? b_qei : b_qsi;
    float* out        = which ? inter_e : inter_s;
    float a0 = bias[t], a1 = bias[t + 256];
    const float* q = q_state + (size_t)b * SN;
    for (int k = 0; k < SN; ++k) {
        const float qk = q[k];
        a0 = fmaf(qk, W[(size_t)k * SN + t], a0);
        a1 = fmaf(qk, W[(size_t)k * SN + t + 256], a1);
    }
    out[(size_t)b * SN + t]       = a0;
    out[(size_t)b * SN + t + 256] = a1;
}

// ---------------------------------------------------------------------------
// Kernel 3/5: fused scoring GEMM.
//   hidden = es[b] (512xK512 bf16) @ W_eff[b] (512x512 bf16) + inter, relu,
//   scores = hidden . w_out, reduced in-block. W_eff folded on the fly:
//     mode 0 (start): W_eff[k,n] = q[k]*W[k,n] + W[512+k,n]
//     mode 1 (end):   W_eff[k,n] = u[k]*W[k,n] + q[k]*W[512+k,n] + W[1024+k,n]
//   WG = 64 rows x 512 cols, 8 waves: wave = (mstrip 0..3) x (col-half 0..1),
//   each wave: 16 WMMA 16x16 tiles, bf16 K-blocks of 32. B fragments are
//   software-pipelined so each WMMA waits only on dscnt<=2, not 0.
// ---------------------------------------------------------------------------
__global__ __launch_bounds__(256) void qa_score_gemm(
    const float* __restrict__ es, const float* __restrict__ q_state,
    const float* __restrict__ inter, const float* __restrict__ W,
    const float* __restrict__ w_out, const int* __restrict__ sptr,
    int mode, float* __restrict__ scores)
{
    __shared__ alignas(16) __bf16 As[64 * 40];    // 64 rows x 32 k, pitch 40
    __shared__ alignas(16) __bf16 Bs[512 * 40];   // 512 n x 32 k (W_eff^T), pitch 40
    __shared__ float ssum[64];

    const int b    = blockIdx.y;
    const int row0 = blockIdx.x * 64;
    const int t    = threadIdx.x;
    const int wave = t >> 5, lane = t & 31;
    const int mstrip = wave >> 1, ch = wave & 1;
    const int lh = lane >> 4, l15 = lane & 15;

    if (t < 64) ssum[t] = 0.f;

    const float* qrow = q_state + (size_t)b * SN;
    const float* urow = es;
    if (mode == 1) {
        const int sp = sptr[b];
        urow = es + ((size_t)b * LSN + sp) * SN;
    }

    v8f acc[16];
    const v8f vz = {0.f, 0.f, 0.f, 0.f, 0.f, 0.f, 0.f, 0.f};
#pragma unroll
    for (int nt = 0; nt < 16; ++nt) acc[nt] = vz;

    // per-lane LDS fragment bases (element indices)
    const int a_base = (mstrip * 16 + l15) * 40 + lh * 16;     // A row for this lane
    const int b_base = (ch * 256 + l15) * 40 + lh * 16;        // B tile-0 row for this lane

    for (int kb = 0; kb < 16; ++kb) {
        __syncthreads();
        // ---- stage A tile: es rows -> bf16 LDS --------------------------
        {
            const int r = t >> 2, sg = t & 3;
            const float* src = es + (((size_t)b * LSN) + row0 + r) * SN + kb * 32 + sg * 8;
            const float4 f0 = *(const float4*)src;
            const float4 f1 = *(const float4*)(src + 4);
            __bf16* dst = &As[r * 40 + sg * 8];
            dst[0] = (__bf16)f0.x; dst[1] = (__bf16)f0.y;
            dst[2] = (__bf16)f0.z; dst[3] = (__bf16)f0.w;
            dst[4] = (__bf16)f1.x; dst[5] = (__bf16)f1.y;
            dst[6] = (__bf16)f1.z; dst[7] = (__bf16)f1.w;
        }
        // ---- stage B tile: per-batch folded weight, transposed ----------
#pragma unroll 4
        for (int i = 0; i < 64; ++i) {
            const int kk = i >> 1;
            const int n  = ((i & 1) << 8) + t;
            const int kg = kb * 32 + kk;
            float val;
            if (mode == 0) {
                val = fmaf(qrow[kg], W[(size_t)kg * SN + n],
                           W[(size_t)(SN + kg) * SN + n]);
            } else {
                val = fmaf(urow[kg], W[(size_t)kg * SN + n],
                      fmaf(qrow[kg], W[(size_t)(SN + kg) * SN + n],
                           W[(size_t)(2 * SN + kg) * SN + n]));
            }
            Bs[n * 40 + kk] = (__bf16)val;
        }
        __syncthreads();

        // ---- A fragment (contiguous k = 16h + s, matching B) -------------
        const v8bf alo = *(const v8bf*)&As[a_base];
        const v8bf ahi = *(const v8bf*)&As[a_base + 8];
        const v16bf a = __builtin_shufflevector(
            alo, ahi, 0, 1, 2, 3, 4, 5, 6, 7, 8, 9, 10, 11, 12, 13, 14, 15);

        // ---- software-pipelined B fragments + WMMA ----------------------
        v8bf b0 = *(const v8bf*)&Bs[b_base];
        v8bf b1 = *(const v8bf*)&Bs[b_base + 8];
#pragma unroll
        for (int nt = 0; nt < 16; ++nt) {
            const v16bf bm = __builtin_shufflevector(
                b0, b1, 0, 1, 2, 3, 4, 5, 6, 7, 8, 9, 10, 11, 12, 13, 14, 15);
            if (nt < 15) {  // prefetch next tile's fragment before the WMMA
                const int nb = b_base + (nt + 1) * 16 * 40;
                b0 = *(const v8bf*)&Bs[nb];
                b1 = *(const v8bf*)&Bs[nb + 8];
            }
            acc[nt] = __builtin_amdgcn_wmma_f32_16x16x32_bf16(
                false, a, false, bm, (short)0, acc[nt], false, false);
        }
    }

    // ---- epilogue: +inter, relu, dot w_out, reduce over 512 columns -----
    float rowsum[8];
#pragma unroll
    for (int r = 0; r < 8; ++r) rowsum[r] = 0.f;
#pragma unroll
    for (int nt = 0; nt < 16; ++nt) {
        const int n = ch * 256 + nt * 16 + l15;
        const float wn  = w_out[n];
        const float inn = inter[(size_t)b * SN + n];
#pragma unroll
        for (int r = 0; r < 8; ++r) {
            float h = acc[nt][r] + inn;
            h = h > 0.f ? h : 0.f;
            rowsum[r] = fmaf(h, wn, rowsum[r]);
        }
    }
#pragma unroll
    for (int r = 0; r < 8; ++r)
        for (int off = 1; off < 16; off <<= 1)
            rowsum[r] += __shfl_xor(rowsum[r], off);
    if (l15 == 0) {
#pragma unroll
        for (int r = 0; r < 8; ++r)
            atomicAdd(&ssum[mstrip * 16 + lh * 8 + r], rowsum[r]);
    }
    __syncthreads();
    if (t < 64) scores[(size_t)b * LSN + row0 + t] = ssum[t];
}

// ---------------------------------------------------------------------------
// Kernel 4/6: apply masks in place + first-occurrence argmax
// ---------------------------------------------------------------------------
__global__ __launch_bounds__(256) void qa_finalize(
    float* __restrict__ scores, const int* __restrict__ slen,
    const int* __restrict__ sptr_in, int* __restrict__ sptr_out,
    float* __restrict__ pred_out, int mode)
{
    __shared__ float bv[256];
    __shared__ int   bi[256];
    const int b = blockIdx.x;
    const int t = threadIdx.x;
    const int sl = slen[b];
    int sp = 0;
    if (mode == 1) sp = sptr_in[b];

    float v0 = scores[(size_t)b * LSN + t];
    float v1 = scores[(size_t)b * LSN + t + 256];
    v0 += (t >= sl) ? NEGV : 0.f;
    v1 += (t + 256 >= sl) ? NEGV : 0.f;
    if (mode == 1) {
        v0 += (t < sp) ? NEGV : 0.f;
        v1 += (t + 256 < sp) ? NEGV : 0.f;
    }
    scores[(size_t)b * LSN + t]       = v0;
    scores[(size_t)b * LSN + t + 256] = v1;

    float mv; int mi;
    if (v1 > v0) { mv = v1; mi = t + 256; } else { mv = v0; mi = t; }
    bv[t] = mv; bi[t] = mi;
    __syncthreads();
    for (int s = 128; s > 0; s >>= 1) {
        if (t < s) {
            if (bv[t + s] > bv[t] || (bv[t + s] == bv[t] && bi[t + s] < bi[t])) {
                bv[t] = bv[t + s];
                bi[t] = bi[t + s];
            }
        }
        __syncthreads();
    }
    if (t == 0) {
        if (sptr_out) sptr_out[b] = bi[0];
        pred_out[b] = (float)bi[0];
    }
}

// ---------------------------------------------------------------------------
extern "C" void kernel_launch(void* const* d_in, const int* in_sizes, int n_in,
                              void* d_out, int out_size, void* d_ws, size_t ws_size,
                              hipStream_t stream) {
    const float* eq    = (const float*)d_in[0];
    const float* es    = (const float*)d_in[1];
    const int*   qlen  = (const int*)d_in[2];
    const int*   slen  = (const int*)d_in[3];
    // d_in[4] correct_start, d_in[5] answer2question, d_in[6] is_eval: unused (eval path)
    const float* W_qa  = (const float*)d_in[7];
    const float* W_qsi = (const float*)d_in[8];
    const float* b_qsi = (const float*)d_in[9];
    const float* W_qs  = (const float*)d_in[10];
    const float* W_ss  = (const float*)d_in[11];
    const float* W_qei = (const float*)d_in[12];
    const float* b_qei = (const float*)d_in[13];
    const float* W_qe  = (const float*)d_in[14];
    const float* W_es  = (const float*)d_in[15];

    float* out          = (float*)d_out;
    float* start_scores = out;
    float* end_scores   = out + BB * LSN;
    float* pred_start   = out + 2 * BB * LSN;
    float* pred_end     = out + 2 * BB * LSN + BB;

    float* ws      = (float*)d_ws;
    float* q_state = ws;
    float* inter_s = ws + BB * SN;
    float* inter_e = ws + 2 * BB * SN;
    int*   sptr    = (int*)(ws + 3 * BB * SN);

    qa_qpool<<<BB, 256, 0, stream>>>(eq, qlen, W_qa, q_state);
    qa_inter<<<dim3(BB, 2), 256, 0, stream>>>(q_state, W_qsi, b_qsi, W_qei, b_qei,
                                              inter_s, inter_e);
    qa_score_gemm<<<dim3(LSN / 64, BB), 256, 0, stream>>>(
        es, q_state, inter_s, W_qs, W_ss, nullptr, 0, start_scores);
    qa_finalize<<<BB, 256, 0, stream>>>(start_scores, slen, nullptr, sptr,
                                        pred_start, 0);
    qa_score_gemm<<<dim3(LSN / 64, BB), 256, 0, stream>>>(
        es, q_state, inter_e, W_qe, W_es, sptr, 1, end_scores);
    qa_finalize<<<BB, 256, 0, stream>>>(end_scores, slen, sptr, nullptr,
                                        pred_end, 1);
}
